// MultiHeadAttention_90065464197691
// MI455X (gfx1250) — compile-verified, matrix operand order checked by reading, not run on hardware
//
#include <hip/hip_runtime.h>

typedef __attribute__((ext_vector_type(16))) __bf16 v16bf;
typedef __attribute__((ext_vector_type(8)))  __bf16 v8bf;
typedef __attribute__((ext_vector_type(8)))  float  v8f;
typedef __attribute__((ext_vector_type(4)))  float  v4f;

#define N_B    8
#define S_LEN  1024
#define T_LEN  1024
#define E_DIM  1024
#define H_NUM  16
#define HD_DIM 64
#define SCALE_F 0.125f   // 1/sqrt(64), folded into Q projection

__device__ __forceinline__ v8bf load8_bf16(const float* p) {
  v4f f0 = *(const v4f*)p;
  v4f f1 = *(const v4f*)(p + 4);
  v8bf r;
  r[0] = (__bf16)f0[0]; r[1] = (__bf16)f0[1];
  r[2] = (__bf16)f0[2]; r[3] = (__bf16)f0[3];
  r[4] = (__bf16)f1[0]; r[5] = (__bf16)f1[1];
  r[6] = (__bf16)f1[2]; r[7] = (__bf16)f1[3];
  return r;
}
__device__ __forceinline__ v8bf load8_bf16(const __bf16* p) {
  return *(const v8bf*)p;
}

// Fragment-major LDS layout (16-bit WMMA A/B layout):
//   element k in [0,32): half = (k>>3)&1, i = ((k>>4)<<3)+(k&7)
//   lane = (row_or_col & 15) + 16*half ; fragment = 16 contiguous bf16 per lane.
//   An 8-run (k%8==0) maps to one lane, consecutive i -> 16B ds_store_b128.

// C(M x Nn) = (X(M x K) @ W^T + bias) * oscale ; W is (Nn x K) row-major.
// Block tile 128x64, BK=32, 128 threads (4 waves), double-buffered LDS.
template <typename Tin, typename Tout>
__global__ __launch_bounds__(128) void gemm_xwt_bias(
    const Tin* __restrict__ X, const float* __restrict__ W,
    const float* __restrict__ bias, Tout* __restrict__ C,
    int M, int K, int Nn, float oscale)
{
  __shared__ __align__(32) __bf16 Af[2][8][32][16];
  __shared__ __align__(32) __bf16 Bf[2][4][32][16];
  const int tid  = threadIdx.x;
  const int wave = tid >> 5;
  const int lane = tid & 31;
  const int half = lane >> 4;
  const int lc   = lane & 15;
  const int m0   = blockIdx.x * 128;
  const int n0   = blockIdx.y * 64;

  auto prefetch = [&](int k0, v8bf va[4], v8bf vb[2]) {
    #pragma unroll
    for (int it = 0; it < 4; ++it) {
      int idx = it * 1024 + tid * 8;
      int r = idx >> 5, c = idx & 31;
      va[it] = load8_bf16(X + (size_t)(m0 + r) * K + (k0 + c));
    }
    #pragma unroll
    for (int it = 0; it < 2; ++it) {
      int idx = it * 1024 + tid * 8;
      int nn = idx >> 5, c = idx & 31;
      vb[it] = load8_bf16(W + (size_t)(n0 + nn) * K + (k0 + c));
    }
  };
  auto stage = [&](int buf, const v8bf va[4], const v8bf vb[2]) {
    #pragma unroll
    for (int it = 0; it < 4; ++it) {
      int idx = it * 1024 + tid * 8;
      int r = idx >> 5, c = idx & 31;
      *(v8bf*)&Af[buf][r >> 4][(r & 15) + 16 * ((c >> 3) & 1)][(c >> 4) << 3] = va[it];
    }
    #pragma unroll
    for (int it = 0; it < 2; ++it) {
      int idx = it * 1024 + tid * 8;
      int nn = idx >> 5, c = idx & 31;
      *(v8bf*)&Bf[buf][nn >> 4][(nn & 15) + 16 * ((c >> 3) & 1)][(c >> 4) << 3] = vb[it];
    }
  };

  v8f acc[2][4] = {};
  v8bf va[4], vb2[2];

  prefetch(0, va, vb2);
  stage(0, va, vb2);
  __syncthreads();

  const int steps = K >> 5;
  for (int s = 0; s < steps; ++s) {
    const int cur = s & 1;
    v16bf a[2], b[4];
    #pragma unroll
    for (int rf = 0; rf < 2; ++rf) a[rf] = *(const v16bf*)Af[cur][wave * 2 + rf][lane];
    #pragma unroll
    for (int nt = 0; nt < 4; ++nt) b[nt] = *(const v16bf*)Bf[cur][nt][lane];

    if (s + 1 < steps) prefetch((s + 1) * 32, va, vb2);   // overlap with WMMAs

    #pragma unroll
    for (int rf = 0; rf < 2; ++rf)
      #pragma unroll
      for (int nt = 0; nt < 4; ++nt)
        acc[rf][nt] = __builtin_amdgcn_wmma_f32_16x16x32_bf16(
            false, a[rf], false, b[nt], (short)0, acc[rf][nt], false, false);

    if (s + 1 < steps) stage(cur ^ 1, va, vb2);
    __syncthreads();
  }

  #pragma unroll
  for (int rf = 0; rf < 2; ++rf)
    #pragma unroll
    for (int nt = 0; nt < 4; ++nt) {
      int col = n0 + nt * 16 + lc;
      float bval = bias[col];
      #pragma unroll
      for (int v = 0; v < 8; ++v) {
        int row = m0 + wave * 32 + rf * 16 + half * 8 + v;
        C[(size_t)row * Nn + col] = (Tout)((acc[rf][nt][v] + bval) * oscale);
      }
    }
}

// Flash attention, causal (Q pre-scaled by 1/sqrt(d) in its projection).
// One workgroup = (batch, head, 64-row block); 4 waves x 16 rows.
// K/V tiles double-buffered in fragment-major LDS.
__global__ __launch_bounds__(128) void flash_attn_bf16(
    const __bf16* __restrict__ Qb, const __bf16* __restrict__ Kb,
    const __bf16* __restrict__ Vb, __bf16* __restrict__ Ob)
{
  __shared__ __align__(32) __bf16 Kf[2][2][4][32][16]; // [buf][kc(d)][nt(t)][lane][i]
  __shared__ __align__(32) __bf16 Vf[2][2][4][32][16]; // [buf][kc(t)][dt(d)][lane][i]
  __shared__ __align__(32) __bf16 Pf[4][2][32][16];    // Q staging then per-wave P
  const int tid  = threadIdx.x;
  const int wave = tid >> 5;
  const int lane = tid & 31;
  const int half = lane >> 4;
  const int lc   = lane & 15;
  const int ib   = blockIdx.x;
  const int h    = blockIdx.y;
  const int n    = blockIdx.z;
  const int row0 = ib * 64;
  const size_t qbase  = ((size_t)n * S_LEN) * E_DIM + (size_t)h * HD_DIM;
  const size_t kvbase = ((size_t)n * T_LEN) * E_DIM + (size_t)h * HD_DIM;

  auto prefetchKV = [&](int t0, v8bf kr[4], v8bf vr[4]) {
    #pragma unroll
    for (int it = 0; it < 4; ++it) {
      int idx = it * 1024 + tid * 8;
      int t = idx >> 6, d = idx & 63;
      const size_t off = kvbase + (size_t)(t0 + t) * E_DIM + d;
      kr[it] = *(const v8bf*)(Kb + off);
      vr[it] = *(const v8bf*)(Vb + off);
    }
  };
  auto stageKV = [&](int buf, const v8bf kr[4], const v8bf vr[4]) {
    #pragma unroll
    for (int it = 0; it < 4; ++it) {
      int idx = it * 1024 + tid * 8;
      int t = idx >> 6, d = idx & 63;
      // K: element (k=d, n=t) -> one lane, 8 consecutive i (b128 store)
      {
        int kc = d >> 5, kp = d & 31;
        *(v8bf*)&Kf[buf][kc][t >> 4][(t & 15) + 16 * ((kp >> 3) & 1)][(kp >> 4) << 3] = kr[it];
      }
      // V: element (k=t, n=d) -> 8 consecutive lanes, fixed i (b16 scatter)
      {
        int kc = t >> 5, kp = t & 31;
        int hh = 16 * ((kp >> 3) & 1);
        int ii = ((kp >> 4) << 3) + (kp & 7);
        #pragma unroll
        for (int j = 0; j < 8; ++j) {
          int dd = d + j;
          Vf[buf][kc][dd >> 4][(dd & 15) + hh][ii] = vr[it][j];
        }
      }
    }
  };

  // stage Q block (64x64, already scaled) in fragment-major layout
  #pragma unroll
  for (int it = 0; it < 4; ++it) {
    int idx = it * 1024 + tid * 8;
    int r = idx >> 6, d = idx & 63;
    v8bf val = *(const v8bf*)(Qb + qbase + (size_t)(row0 + r) * E_DIM + d);
    int kc = d >> 5, kp = d & 31;
    *(v8bf*)&Pf[r >> 4][kc][(r & 15) + 16 * ((kp >> 3) & 1)][(kp >> 4) << 3] = val;
  }
  __syncthreads();
  v16bf qa[2];
  #pragma unroll
  for (int kc = 0; kc < 2; ++kc) qa[kc] = *(const v16bf*)Pf[wave][kc][lane];
  __syncthreads();

  const float NEG_INF = -__builtin_inff();
  float mrow[8], lrow[8];
  #pragma unroll
  for (int v = 0; v < 8; ++v) { mrow[v] = NEG_INF; lrow[v] = 0.f; }
  v8f oacc[4] = {};

  v8bf kr[4], vr[4];
  prefetchKV(0, kr, vr);
  stageKV(0, kr, vr);
  __syncthreads();

  for (int jt = 0; jt <= ib; ++jt) {          // causal tiles only
    const int cur = jt & 1;
    const bool have_next = jt < ib;
    if (have_next) prefetchKV((jt + 1) * 64, kr, vr);   // overlap with compute

    // S = Q @ K^T  (16 rows x 64 cols per wave), Q pre-scaled
    v8f sc[4] = {};
    #pragma unroll
    for (int kc = 0; kc < 2; ++kc) {
      #pragma unroll
      for (int nt = 0; nt < 4; ++nt) {
        v16bf b = *(const v16bf*)Kf[cur][kc][nt][lane];
        sc[nt] = __builtin_amdgcn_wmma_f32_16x16x32_bf16(
            false, qa[kc], false, b, (short)0, sc[nt], false, false);
      }
    }

    // causal mask only on the diagonal tile (uniform branch)
    if (jt == ib) {
      const int t0 = jt * 64;
      #pragma unroll
      for (int nt = 0; nt < 4; ++nt) {
        int tg = t0 + nt * 16 + lc;
        #pragma unroll
        for (int v = 0; v < 8; ++v) {
          int rg = row0 + wave * 16 + half * 8 + v;
          if (tg > rg) sc[nt][v] = NEG_INF;
        }
      }
    }

    // online softmax (row stats replicated across the 16 lanes of each half)
    #pragma unroll
    for (int v = 0; v < 8; ++v) {
      float tm = fmaxf(fmaxf(sc[0][v], sc[1][v]), fmaxf(sc[2][v], sc[3][v]));
      #pragma unroll
      for (int off = 1; off < 16; off <<= 1)
        tm = fmaxf(tm, __shfl_xor(tm, off, 32));
      float mnew = fmaxf(mrow[v], tm);
      float corr = __expf(mrow[v] - mnew);
      mrow[v] = mnew;
      float rs = 0.f;
      #pragma unroll
      for (int nt = 0; nt < 4; ++nt) {
        float p = __expf(sc[nt][v] - mnew);
        sc[nt][v] = p;
        rs += p;
      }
      #pragma unroll
      for (int off = 1; off < 16; off <<= 1)
        rs += __shfl_xor(rs, off, 32);
      lrow[v] = lrow[v] * corr + rs;
      #pragma unroll
      for (int nt = 0; nt < 4; ++nt) oacc[nt][v] *= corr;
    }

    // P (C-layout) -> wave-private fragment-major staging:
    //   element (m = half*8+v, k = nt*16+lc):
    //   kc = nt>>1 ; lane' = m + 16*((lc>>3)&1) ; i = ((nt&1)<<3)+(lc&7)
    #pragma unroll
    for (int nt = 0; nt < 4; ++nt) {
      int kc = nt >> 1;
      int ii = ((nt & 1) << 3) + (lc & 7);
      int hh = 16 * ((lc >> 3) & 1);
      #pragma unroll
      for (int v = 0; v < 8; ++v)
        Pf[wave][kc][(half * 8 + v) + hh][ii] = (__bf16)sc[nt][v];
    }

    // O += P @ V
    #pragma unroll
    for (int kc = 0; kc < 2; ++kc) {
      v16bf pa = *(const v16bf*)Pf[wave][kc][lane];
      #pragma unroll
      for (int dt = 0; dt < 4; ++dt) {
        v16bf vb = *(const v16bf*)Vf[cur][kc][dt][lane];
        oacc[dt] = __builtin_amdgcn_wmma_f32_16x16x32_bf16(
            false, pa, false, vb, (short)0, oacc[dt], false, false);
      }
    }

    if (have_next) stageKV(cur ^ 1, kr, vr);
    __syncthreads();
  }

  // O = acc / l
  #pragma unroll
  for (int dt = 0; dt < 4; ++dt) {
    #pragma unroll
    for (int v = 0; v < 8; ++v) {
      int row = row0 + wave * 16 + half * 8 + v;
      int d = dt * 16 + lc;
      Ob[qbase + (size_t)row * E_DIM + d] = (__bf16)(oacc[dt][v] / lrow[v]);
    }
  }
}

extern "C" void kernel_launch(void* const* d_in, const int* in_sizes, int n_in,
                              void* d_out, int out_size, void* d_ws, size_t ws_size,
                              hipStream_t stream) {
  (void)in_sizes; (void)n_in; (void)out_size; (void)ws_size;
  const float* query = (const float*)d_in[0];
  const float* key   = (const float*)d_in[1];
  const float* value = (const float*)d_in[2];
  // d_in[3] = attn_mask (causal tril) — implemented analytically in-kernel
  const float* Wq = (const float*)d_in[4];
  const float* bq = (const float*)d_in[5];
  const float* Wk = (const float*)d_in[6];
  const float* bk = (const float*)d_in[7];
  const float* Wv = (const float*)d_in[8];
  const float* bv = (const float*)d_in[9];
  const float* Wp = (const float*)d_in[10];
  const float* bp = (const float*)d_in[11];

  const size_t elems = (size_t)N_B * S_LEN * E_DIM;   // 8 Mi
  __bf16* qb = (__bf16*)d_ws;                         // 16 MB each
  __bf16* kb = qb + elems;
  __bf16* vb = kb + elems;
  __bf16* ab = vb + elems;

  const int M = N_B * S_LEN;                          // 8192
  dim3 blk(128);
  dim3 ggrid(M / 128, E_DIM / 64);                    // 64 x 16

  gemm_xwt_bias<float, __bf16><<<ggrid, blk, 0, stream>>>(query, Wq, bq, qb, M, E_DIM, E_DIM, SCALE_F);
  gemm_xwt_bias<float, __bf16><<<ggrid, blk, 0, stream>>>(key,   Wk, bk, kb, M, E_DIM, E_DIM, 1.0f);
  gemm_xwt_bias<float, __bf16><<<ggrid, blk, 0, stream>>>(value, Wv, bv, vb, M, E_DIM, E_DIM, 1.0f);

  flash_attn_bf16<<<dim3(S_LEN / 64, H_NUM, N_B), blk, 0, stream>>>(qb, kb, vb, ab);

  gemm_xwt_bias<__bf16, float><<<ggrid, blk, 0, stream>>>(ab, Wp, bp, (float*)d_out, M, E_DIM, E_DIM, 1.0f);
}
